// dLSTM_1202590843593
// MI455X (gfx1250) — compile-verified, matrix operand order checked by reading, not run on hardware
//
#include <hip/hip_runtime.h>

// ---------------------------------------------------------------------------
// Types for CDNA5 WMMA (gfx1250, wave32)
// ---------------------------------------------------------------------------
typedef __attribute__((ext_vector_type(16))) __bf16 v16bf;
typedef __attribute__((ext_vector_type(8)))  __bf16 v8bf;
typedef __attribute__((ext_vector_type(8)))  float  v8f;
typedef __attribute__((ext_vector_type(4)))  unsigned int u32x4;
typedef __attribute__((ext_vector_type(4)))  int          i32x4;
typedef __attribute__((ext_vector_type(8)))  int          i32x8;

// Use the CDNA5 Tensor Data Mover for global->LDS A staging (1 = on).
#ifndef USE_TDM
#define USE_TDM 1
#endif

// Sizes fixed by the reference problem
// L=2, B=64, H=2048, IN=1024, gates 4H=8192, concat K=4096
#define HID   2048
#define BATCH 64
#define NIN   1024

// Fragment-buffer geometry (bf16 halves):
//  B (weights):  [nTile][kStep][lane32][j16]   gates: nTiles=512, kSteps=128
//  A (acts/h):   [mTile4][kStep64][lane32][j16]
#define GATE_B_HALVES (512LL * 128 * 32 * 16)   // 33,554,432 per layer
#define REC_B_HALVES  (64LL * 64 * 32 * 16)     //  2,097,152
#define A_HALVES      (4LL * 64 * 32 * 16)      //    131,072 (= 64x2048)

// ---------------------------------------------------------------------------
// Helpers
// ---------------------------------------------------------------------------
__device__ __forceinline__ __bf16 f2bf(float f) {
    unsigned u = __builtin_bit_cast(unsigned, f);
    unsigned r = u + 0x7FFFu + ((u >> 16) & 1u);   // round-to-nearest-even
    unsigned short h = (unsigned short)(r >> 16);
    return __builtin_bit_cast(__bf16, h);
}

__device__ __forceinline__ float sigmoidf(float x) {
    return 1.0f / (1.0f + __expf(-x));
}

// ISA 16-bit A-matrix 16x32 K mapping: k = (v<4?0:16) + 8*half + 2*(v&3) + p
__device__ __forceinline__ int frag_k(int half, int v, int p) {
    return ((v & 4) << 2) + (half << 3) + ((v & 3) << 1) + p;
}

// ---------------------------------------------------------------------------
// One-time packing kernels (run per launch; cheap vs 128-step loop)
// ---------------------------------------------------------------------------
__global__ void pack_b_kernel(const float* __restrict__ W0,
                              const float* __restrict__ W1,
                              __bf16* __restrict__ out,
                              int kSteps, int kSplit, long long total) {
    long long idx = (long long)blockIdx.x * blockDim.x + threadIdx.x;
    if (idx >= total) return;
    int j    = (int)(idx & 15);
    int lane = (int)((idx >> 4) & 31);
    long long rest = idx >> 9;
    int ks = (int)(rest % kSteps);
    int nT = (int)(rest / kSteps);
    int n    = (nT << 4) + (lane & 15);
    int half = lane >> 4;
    int p = j & 1, v = j >> 1;
    int k = (ks << 5) + frag_k(half, v, p);
    float w = (k < kSplit) ? W0[(long long)n * HID + k]
                           : W1[(long long)n * HID + (k - kSplit)];
    out[idx] = f2bf(w);
}

__global__ void pack_a_kernel(const float* __restrict__ h,
                              __bf16* __restrict__ out) {
    int idx = blockIdx.x * blockDim.x + threadIdx.x;   // < 131072
    int j    = idx & 15;
    int lane = (idx >> 4) & 31;
    int ks   = (idx >> 9) & 63;
    int mT   = idx >> 15;
    int m    = lane & 15;
    int half = lane >> 4;
    int p = j & 1, v = j >> 1;
    int b = (mT << 4) + m;
    int k = (ks << 5) + frag_k(half, v, p);
    out[idx] = f2bf(h[b * HID + k]);
}

__global__ void bias_sum_kernel(const float* __restrict__ bih,
                                const float* __restrict__ bhh,
                                float* __restrict__ out, int n) {
    int i = blockIdx.x * blockDim.x + threadIdx.x;
    if (i < n) out[i] = bih[i] + bhh[i];
}

__global__ void copy_f32_kernel(const float* __restrict__ src,
                                float* __restrict__ dst, int n) {
    int i = blockIdx.x * blockDim.x + threadIdx.x;
    if (i < n) dst[i] = src[i];
}

__global__ void zero_bf16_kernel(__bf16* __restrict__ dst, int n) {
    int i = blockIdx.x * blockDim.x + threadIdx.x;
    if (i < n) dst[i] = f2bf(0.0f);
}

// ---------------------------------------------------------------------------
// TDM helper: issue one tensor_load_to_lds copying a 2D tile
//   4 rows (mTiles) x 16384 bytes, row stride 65536 bytes  ->  LDS (contig).
// Descriptor fields per CDNA5 ISA.
// ---------------------------------------------------------------------------
#if USE_TDM
__device__ __forceinline__ void tdm_load_chunk(const __bf16* gsrc, unsigned ldsByteAddr) {
    unsigned long long ga = (unsigned long long)(size_t)gsrc;  // 57-bit global addr
    u32x4 g0;
    g0[0] = 1u;                                  // count=1, is_restore=0, gather=0
    g0[1] = ldsByteAddr;                         // lds_addr [63:32]
    g0[2] = (unsigned)(ga & 0xFFFFFFFFu);        // global_addr [95:64]
    g0[3] = (unsigned)((ga >> 32) & 0x01FFFFFFu) // global_addr [120:96]
            | (2u << 30);                        // type=2 ("image") [127:126]

    // data_size=3 (8B units). Tile: dim0=2048 units (16KB), dim1=4 rows.
    // Row stride (tensor_dim0_stride) = 8192 units (64KB). Large tensor dims
    // so the OOB clamp never triggers.
    i32x8 g1;
    g1[0] = (3 << 16);                           // workgroup_mask=0, data_size=3
    g1[1] = 0;                                   // abar_addr=0, tensor_dim0[15:0]=0
    g1[2] = (int)((4u << 16) | 0x10u);           // tensor_dim0=0x100000, tensor_dim1=4
    g1[3] = (int)(2048u << 16);                  // tile_dim0=2048
    g1[4] = 4;                                   // tile_dim1=4, tile_dim2=0
    g1[5] = 8192;                                // tensor_dim0_stride[31:0]
    g1[6] = 0;                                   // stride msbs / dim1_stride lo
    g1[7] = 0;
    i32x4 gz = {0, 0, 0, 0};
#if defined(__clang_major__) && (__clang_major__ >= 23)
    i32x8 gz8 = {0, 0, 0, 0, 0, 0, 0, 0};
    __builtin_amdgcn_tensor_load_to_lds(g0, g1, gz, gz, gz8, 0);
#else
    __builtin_amdgcn_tensor_load_to_lds(g0, g1, gz, gz, 0);
#endif
}
#endif

// ---------------------------------------------------------------------------
// LSTM layer step: gates = [x|h] @ Wcat^T + bias ; pointwise update of c,h.
// Grid: 16 blocks x 256 threads (8 waves). Wave w owns hidden tile
// hTile = blockIdx.x*8 + w, computing i/f/g/o for all 64 batch rows
// (16 f32 accumulators / 128 VGPRs). A fragments are staged per 16-K-step
// chunk (64 KB) in LDS — via the Tensor Data Mover (TENSORcnt) — and shared
// by all 8 waves; B (weights) streams from the 192 MB L2.
// LDS layout (TDM row order): [mT][ksl][lane][j], 8192 halves per mT.
// ---------------------------------------------------------------------------
#define KCHUNK 16
#define LDS_HALVES (KCHUNK * 4 * 512)   // 32768 halves = 64 KB

__global__ void __launch_bounds__(256)
lstm_layer_kernel(const __bf16* __restrict__ Ax,    // x part, A-frag layout
                  const __bf16* __restrict__ Ah,    // h part, A-frag layout
                  const __bf16* __restrict__ Bp,    // packed [Wih;Whh]^T
                  const float*  __restrict__ bias,  // [8192] = b_ih + b_hh
                  float*        __restrict__ cst,   // [64][2048] fp32 cell
                  __bf16*       __restrict__ Hout)  // new h, A-frag layout
{
    __shared__ __bf16 Asmem[LDS_HALVES];

    const int lane  = threadIdx.x & 31;
    const int wave  = threadIdx.x >> 5;
    const int hTile = __builtin_amdgcn_readfirstlane((blockIdx.x << 3) + wave);
    const int laneOff = lane << 4;

    // Scalar (SGPR) per-gate B base pointers -> saddr-form global loads.
    const __bf16* bbase0 = Bp + (((size_t)((0 * 128 + hTile) * 128)) << 9);
    const __bf16* bbase1 = Bp + (((size_t)((1 * 128 + hTile) * 128)) << 9);
    const __bf16* bbase2 = Bp + (((size_t)((2 * 128 + hTile) * 128)) << 9);
    const __bf16* bbase3 = Bp + (((size_t)((3 * 128 + hTile) * 128)) << 9);

    v8f zero = {};
    v8f acc[4][4];                                   // [gate][mTile]
#pragma unroll
    for (int g = 0; g < 4; ++g)
#pragma unroll
        for (int m = 0; m < 4; ++m) acc[g][m] = zero;

    for (int chunk = 0; chunk < 8; ++chunk) {
        const __bf16* asrc = ((chunk < 4) ? Ax : Ah) + (size_t)(chunk & 3) * (KCHUNK * 512);
        __syncthreads();    // previous chunk fully consumed
#if USE_TDM
        if (wave == 0) {
            tdm_load_chunk(asrc, (unsigned)(size_t)&Asmem[0]);
            __builtin_amdgcn_s_wait_tensorcnt(0);
        }
#else
        // Cooperative copy fallback: [mT][ksl][lane][j] in LDS.
#pragma unroll
        for (int i = 0; i < 16; ++i) {
            const int u   = (i << 8) + threadIdx.x;   // 16-byte unit, 0..4095
            const int j8  = u & 1;
            const int lns = (u >> 1) & 31;
            const int ksl = (u >> 6) & 15;
            const int mT  = u >> 10;
            *(v8bf*)(Asmem + (size_t)u * 8) =
                *(const v8bf*)(asrc + (size_t)mT * 32768 +
                               (size_t)ksl * 512 + lns * 16 + j8 * 8);
        }
#endif
        __syncthreads();

        // Prefetch next chunk's B fragments into WGP$ (global_prefetch_b8).
        if (chunk < 7) {
            const int ksn = (chunk + 1) * KCHUNK;
            __builtin_prefetch(bbase0 + ((size_t)ksn << 9) + laneOff, 0, 0);
            __builtin_prefetch(bbase1 + ((size_t)ksn << 9) + laneOff, 0, 0);
            __builtin_prefetch(bbase2 + ((size_t)ksn << 9) + laneOff, 0, 0);
            __builtin_prefetch(bbase3 + ((size_t)ksn << 9) + laneOff, 0, 0);
        }

        // ---- Compute: 16 K-steps x 16 WMMA ----
        for (int ksl = 0; ksl < KCHUNK; ++ksl) {
            const size_t kOff = ((size_t)(chunk * KCHUNK + ksl) << 9) + laneOff;
            v16bf bfr[4];
            bfr[0] = *(const v16bf*)(bbase0 + kOff);
            bfr[1] = *(const v16bf*)(bbase1 + kOff);
            bfr[2] = *(const v16bf*)(bbase2 + kOff);
            bfr[3] = *(const v16bf*)(bbase3 + kOff);
            // A fragments from LDS: [mT][ksl][lane][j]
            const __bf16* ab = Asmem + (size_t)ksl * 512 + laneOff;
            v16bf a0 = *(const v16bf*)(ab + 0 * 8192);
            v16bf a1 = *(const v16bf*)(ab + 1 * 8192);
            v16bf a2 = *(const v16bf*)(ab + 2 * 8192);
            v16bf a3 = *(const v16bf*)(ab + 3 * 8192);
#pragma unroll
            for (int g = 0; g < 4; ++g) {
                acc[g][0] = __builtin_amdgcn_wmma_f32_16x16x32_bf16(
                    false, a0, false, bfr[g], (short)0, acc[g][0], false, false);
                acc[g][1] = __builtin_amdgcn_wmma_f32_16x16x32_bf16(
                    false, a1, false, bfr[g], (short)0, acc[g][1], false, false);
                acc[g][2] = __builtin_amdgcn_wmma_f32_16x16x32_bf16(
                    false, a2, false, bfr[g], (short)0, acc[g][2], false, false);
                acc[g][3] = __builtin_amdgcn_wmma_f32_16x16x32_bf16(
                    false, a3, false, bfr[g], (short)0, acc[g][3], false, false);
            }
        }
    }

    // Pointwise LSTM update. C/D layout: VGPR r, lanes 0-15 -> M=r, N=lane;
    // lanes 16-31 -> M=r+8. Gate order i,f,g,o across the 8192 columns.
    const int col  = (hTile << 4) + (lane & 15);     // hidden unit 0..2047
    const int half = lane >> 4;
    const float bi = bias[col];
    const float bf = bias[HID + col];
    const float bg = bias[2 * HID + col];
    const float bo = bias[3 * HID + col];

    const int kk = col & 31, kS = col >> 5;
    const int pa = kk & 1, hfa = (kk >> 3) & 1;
    const int va = ((kk >> 4) << 2) + ((kk >> 1) & 3);
    const int ja = (va << 1) + pa;

#pragma unroll
    for (int m = 0; m < 4; ++m) {
#pragma unroll
        for (int r = 0; r < 8; ++r) {
            const int row = (m << 4) + r + (half << 3);   // batch row
            const float gi = acc[0][m][r] + bi;
            const float gf = acc[1][m][r] + bf;
            const float gg = acc[2][m][r] + bg;
            const float go = acc[3][m][r] + bo;
            const float cv = cst[row * HID + col];
            const float cn = sigmoidf(gf) * cv + sigmoidf(gi) * tanhf(gg);
            const float hn = sigmoidf(go) * tanhf(cn);
            cst[row * HID + col] = cn;
            const int laneA = (hfa << 4) + (row & 15);
            const int mT    = row >> 4;
            Hout[((((size_t)(mT * 64 + kS)) << 5) + laneA) * 16 + ja] = f2bf(hn);
        }
    }
}

// ---------------------------------------------------------------------------
// Reconstruction: out = h1 @ W_rec^T + b_rec, written time-flipped.
// Grid: 16 blocks x 128 threads (64 waves), wave owns one 16-col output tile.
// ---------------------------------------------------------------------------
__global__ void __launch_bounds__(128)
recons_kernel(const __bf16* __restrict__ Ah,   // h1, A-frag layout
              const __bf16* __restrict__ Bp,   // packed W_rec^T
              const float*  __restrict__ brec, // [1024]
              float*        __restrict__ out)  // [64][1024] slice of d_out
{
    const int lane  = threadIdx.x & 31;
    const int wave  = threadIdx.x >> 5;
    const int nTile = __builtin_amdgcn_readfirstlane((blockIdx.x << 2) + wave);
    const int laneOff = lane << 4;

    const __bf16* bbase = Bp + (((size_t)(nTile * 64)) << 9);

    v8f zero = {};
    v8f acc[4];
#pragma unroll
    for (int m = 0; m < 4; ++m) acc[m] = zero;

    for (int ks = 0; ks < 64; ++ks) {
        const __bf16* as = Ah + ((size_t)ks << 9);
        v16bf b = *(const v16bf*)(bbase + ((size_t)ks << 9) + laneOff);
        v16bf a0 = *(const v16bf*)(as + 0 * 32768 + laneOff);
        v16bf a1 = *(const v16bf*)(as + 1 * 32768 + laneOff);
        v16bf a2 = *(const v16bf*)(as + 2 * 32768 + laneOff);
        v16bf a3 = *(const v16bf*)(as + 3 * 32768 + laneOff);
        acc[0] = __builtin_amdgcn_wmma_f32_16x16x32_bf16(
            false, a0, false, b, (short)0, acc[0], false, false);
        acc[1] = __builtin_amdgcn_wmma_f32_16x16x32_bf16(
            false, a1, false, b, (short)0, acc[1], false, false);
        acc[2] = __builtin_amdgcn_wmma_f32_16x16x32_bf16(
            false, a2, false, b, (short)0, acc[2], false, false);
        acc[3] = __builtin_amdgcn_wmma_f32_16x16x32_bf16(
            false, a3, false, b, (short)0, acc[3], false, false);
    }

    const int col  = (nTile << 4) + (lane & 15);
    const int half = lane >> 4;
    const float bb = brec[col];
#pragma unroll
    for (int m = 0; m < 4; ++m)
#pragma unroll
        for (int r = 0; r < 8; ++r) {
            const int row = (m << 4) + r + (half << 3);
            out[row * NIN + col] = acc[m][r] + bb;
        }
}

// ---------------------------------------------------------------------------
// Host driver
// ---------------------------------------------------------------------------
extern "C" void kernel_launch(void* const* d_in, const int* in_sizes, int n_in,
                              void* d_out, int out_size, void* d_ws, size_t ws_size,
                              hipStream_t stream) {
    const float* h0   = (const float*)d_in[1];
    const float* c0   = (const float*)d_in[2];
    const float* Wih  = (const float*)d_in[3];
    const float* Whh  = (const float*)d_in[4];
    const float* bih  = (const float*)d_in[5];
    const float* bhh  = (const float*)d_in[6];
    const float* Wrec = (const float*)d_in[7];
    const float* brec = (const float*)d_in[8];
    float* out = (float*)d_out;

    const int T = out_size / (BATCH * NIN);   // seq_len (=128)

    char* ws = (char*)d_ws;
    size_t off = 0;
    auto carve = [&](size_t bytes) -> void* {
        void* p = ws + off;
        off = (off + bytes + 255) & ~(size_t)255;
        return p;
    };
    __bf16* Bpack  = (__bf16*)carve(2 * GATE_B_HALVES * sizeof(__bf16)); // 128 MB
    __bf16* Rpack  = (__bf16*)carve(REC_B_HALVES * sizeof(__bf16));      //   4 MB
    float*  bias   = (float*)carve(2 * 4 * HID * sizeof(float));
    __bf16* H0buf0 = (__bf16*)carve(A_HALVES * sizeof(__bf16));
    __bf16* H0buf1 = (__bf16*)carve(A_HALVES * sizeof(__bf16));
    __bf16* H1buf0 = (__bf16*)carve(A_HALVES * sizeof(__bf16));
    __bf16* H1buf1 = (__bf16*)carve(A_HALVES * sizeof(__bf16));
    __bf16* Zbuf   = (__bf16*)carve(A_HALVES * sizeof(__bf16));
    float*  cbuf   = (float*)carve(2 * BATCH * HID * sizeof(float));
    if (off > ws_size) return;  // insufficient workspace

    __bf16* H0buf[2] = {H0buf0, H0buf1};
    __bf16* H1buf[2] = {H1buf0, H1buf1};

    const int TPB = 256;

    // --- One-time packing: bf16 weights resident in 192 MB L2 thereafter ---
    {
        const long long tot = GATE_B_HALVES;
        const int nb = (int)((tot + TPB - 1) / TPB);
        pack_b_kernel<<<nb, TPB, 0, stream>>>(Wih, Whh, Bpack, 128, HID, tot);
        pack_b_kernel<<<nb, TPB, 0, stream>>>(Wih + (size_t)4 * HID * HID,
                                              Whh + (size_t)4 * HID * HID,
                                              Bpack + GATE_B_HALVES, 128, HID, tot);
        const long long rtot = REC_B_HALVES;
        pack_b_kernel<<<(int)((rtot + TPB - 1) / TPB), TPB, 0, stream>>>(
            Wrec, Wrec, Rpack, 64, 2 * HID /*never split*/, rtot);
    }
    pack_a_kernel<<<(int)(A_HALVES / TPB), TPB, 0, stream>>>(h0, H0buf[0]);
    pack_a_kernel<<<(int)(A_HALVES / TPB), TPB, 0, stream>>>(h0 + BATCH * HID, H1buf[0]);
    bias_sum_kernel<<<(2 * 4 * HID) / TPB, TPB, 0, stream>>>(bih, bhh, bias, 2 * 4 * HID);
    copy_f32_kernel<<<(2 * BATCH * HID) / TPB, TPB, 0, stream>>>(c0, cbuf, 2 * BATCH * HID);
    zero_bf16_kernel<<<(int)(A_HALVES / TPB), TPB, 0, stream>>>(Zbuf, (int)A_HALVES);

    // --- Recurrence: 3 launches per step, weights served from L2 ---
    for (int t = 0; t < T; ++t) {
        const int rd = t & 1, wr = rd ^ 1;
        const __bf16* x0 = (t == 0) ? Zbuf : H1buf[rd];
        lstm_layer_kernel<<<16, 256, 0, stream>>>(
            x0, H0buf[rd], Bpack, bias, cbuf, H0buf[wr]);
        lstm_layer_kernel<<<16, 256, 0, stream>>>(
            H0buf[wr], H1buf[rd], Bpack + GATE_B_HALVES, bias + 4 * HID,
            cbuf + BATCH * HID, H1buf[wr]);
        recons_kernel<<<16, 128, 0, stream>>>(
            H1buf[wr], Rpack, brec, out + (size_t)(T - 1 - t) * BATCH * NIN);
    }
}
